// DifferentiableVectorization_27839978013480
// MI455X (gfx1250) — matrix operand order repaired and checked by reading, not compiled, not run on hardware
//
#include <hip/hip_runtime.h>
#include <hip/hip_bf16.h>

// CDNA5 (gfx1250, wave32) DifferentiableVectorization.
// All dense layers: v_wmma_f32_16x16x32_f16, f32 accumulate.
// Weights/activations are pre-converted to f16 and zero-padded to
// (K mult 32, N mult 64) so the GEMM hot loop is completely branch-free:
// coalesced vector global loads -> LDS -> ds_load_b128 fragments -> WMMA.

typedef __attribute__((ext_vector_type(16))) _Float16 v16h;
typedef __attribute__((ext_vector_type(8)))  _Float16 v8h;
typedef __attribute__((ext_vector_type(4)))  _Float16 v4h;
typedef __attribute__((ext_vector_type(8)))  float    v8f;

#define NB      4
#define FDIM    256
#define HH      256
#define WW      256
#define MAX_P   20
#define MAX_N   50
#define QPTS    (MAX_P * MAX_N)   // 1000
#define NQ      (NB * QPTS)       // 4000
#define SCALEF  0.08f

#define ACT_RELU    1
#define ACT_SIGMOID 2
#define ACT_TANH    3

// ---------------------------------------------------------------------------
// f32 [M,K] -> f16 [M2,K2] zero-padded
// ---------------------------------------------------------------------------
__global__ void cvt_pad_kernel(const float* __restrict__ src, _Float16* __restrict__ dst,
                               int M, int K, int M2, int K2) {
    int i = blockIdx.x * blockDim.x + threadIdx.x;
    if (i >= M2 * K2) return;
    int m = i / K2, k = i % K2;
    int mc = m < M ? m : M - 1;
    int kc = k < K ? k : K - 1;
    float f = src[(size_t)mc * K + kc];
    dst[i] = (_Float16)((m < M && k < K) ? f : 0.f);
}

__global__ void fill_zero_h(_Float16* __restrict__ p, int n) {
    int i = blockIdx.x * blockDim.x + threadIdx.x;
    if (i < n) p[i] = (_Float16)0.f;
}

// ---------------------------------------------------------------------------
// AdaptiveAvgPool2d(8) on p4 [4,256,64,64] -> f16 flat rows 0..3 of [16,16384]
// ---------------------------------------------------------------------------
__global__ void pool_kernel(const float* __restrict__ p4, _Float16* __restrict__ out) {
    int idx = blockIdx.x * blockDim.x + threadIdx.x;     // 65536
    if (idx >= NB * FDIM * 64) return;
    int b   = idx / (FDIM * 64);
    int rem = idx % (FDIM * 64);
    int c   = rem / 64;
    int oy  = (rem % 64) / 8;
    int ox  = rem % 8;
    const float* base = p4 + (((size_t)(b * FDIM + c) * 64 + oy * 8) * 64) + ox * 8;
    float s = 0.f;
#pragma unroll
    for (int i = 0; i < 8; ++i)
#pragma unroll
        for (int j = 0; j < 8; ++j)
            s += base[i * 64 + j];
    out[(size_t)b * 16384 + rem] = (_Float16)(s * (1.f / 64.f));
}

// ---------------------------------------------------------------------------
// WMMA GEMM on padded f16 operands. A2:[M2,K2], W2:[K2,N2]; K2%32==0, N2%64==0,
// grid = (N2/64, M2/16), block = 128 (4 waves, one 16x16 N-subtile each).
// Outputs: Cf16 [M2,N2] padded (mask-to-zero, branch-free) and/or Cf32 [M,N].
// Only the tid==0 prefetch is conditional in the k-loop -> no saveexec around loads.
// ---------------------------------------------------------------------------
__global__ void wmma_gemm_kernel(const _Float16* __restrict__ A2,
                                 const _Float16* __restrict__ W2,
                                 const float* __restrict__ bias,   // [N]
                                 float* __restrict__ Cf32,         // [M,N] or null
                                 _Float16* __restrict__ Cf16,      // [M2,N2] or null
                                 int M, int N, int K2, int N2, int act) {
    __shared__ __align__(16) _Float16 As[16 * 32];   // 1 KB
    __shared__ __align__(16) _Float16 Bs[32 * 64];   // 4 KB

    const int tid    = threadIdx.x;        // 0..127
    const int lane   = tid & 31;
    const int wid    = tid >> 5;
    const int half   = lane >> 4;
    const int l16    = lane & 15;
    const int tileM  = blockIdx.y * 16;
    const int tileN0 = blockIdx.x * 64;

    // Per-thread fill addresses (loop-invariant parts)
    const int arow = tid >> 3;             // 0..15
    const int aseg = tid & 7;              // 0..7 (x4 halfs)
    const int brow = tid >> 2;             // 0..31
    const int bseg = tid & 3;              // 0..3 (x16 halfs)

    v8f acc = {};

    for (int k0 = 0; k0 < K2; k0 += 32) {
        __syncthreads();   // previous iteration's LDS reads complete

        // A tile: 16x32 halfs, one v4h (8B) per thread, coalesced
        {
            const v4h* src = (const v4h*)(A2 + (size_t)(tileM + arow) * K2 + k0) + aseg;
            *((v4h*)(As + (arow << 5)) + aseg) = *src;
        }
        // B tile: 32x64 halfs, two v8h (16B) per thread, coalesced
        {
            const v8h* src = (const v8h*)(W2 + (size_t)(k0 + brow) * N2 + tileN0 + (bseg << 4));
            v8h* dst = (v8h*)(Bs + (brow << 6) + (bseg << 4));
            dst[0] = src[0];
            dst[1] = src[1];
        }
        if (tid == 0 && k0 + 32 < K2)
            __builtin_prefetch(W2 + (size_t)(k0 + 32) * N2 + tileN0, 0, 1);

        __syncthreads();

        // A-frag: row m = l16; runs k = h*8..+7 and 16+h*8..+7 (ISA 7.12.2)
        const v8h* Arow = (const v8h*)(As + (l16 << 5));
        v8h a_lo = Arow[half];
        v8h a_hi = Arow[2 + half];
        v16h a = __builtin_shufflevector(a_lo, a_hi,
                     0, 1, 2, 3, 4, 5, 6, 7, 8, 9, 10, 11, 12, 13, 14, 15);

        // B-frag: col n = l16 of this wave's subtile; k = h*8..+7 and 16+h*8..+7
        const _Float16* Bcol = Bs + (wid << 4) + l16;
        v16h b;
#pragma unroll
        for (int i = 0; i < 8; ++i) {
            b[i]     = Bcol[((half << 3) + i) << 6];
            b[8 + i] = Bcol[(16 + (half << 3) + i) << 6];
        }

        acc = __builtin_amdgcn_wmma_f32_16x16x32_f16(
                  false, a, false, b, (short)0, acc, false, false);
    }

    // Store. C layout: lane -> col n = l16 (+wave subtile); acc[r] -> row r+half*8
    int n  = tileN0 + (wid << 4) + l16;
    int nc = n < N ? n : N - 1;
    float bv = bias[nc];
    bv = n < N ? bv : 0.f;
#pragma unroll
    for (int r = 0; r < 8; ++r) {
        int m = tileM + r + (half << 3);
        float v = acc[r] + bv;
        if      (act == ACT_RELU)    v = fmaxf(v, 0.f);
        else if (act == ACT_SIGMOID) v = 1.f / (1.f + __expf(-v));
        else if (act == ACT_TANH)    v = tanhf(v);
        bool in = (m < M) & (n < N);
        if (Cf16) Cf16[(size_t)m * N2 + n] = (_Float16)(in ? v : 0.f);
        if (Cf32 && in) Cf32[(size_t)m * N + n] = v;
    }
}

// ---------------------------------------------------------------------------
// Bilinear grid_sample (border, align_corners=False) from p2 [B,C,H,W]
// -> f16 padded inp [NQ, 288] = [256 ch | x | y | 30 zeros]
// ---------------------------------------------------------------------------
__global__ void sample_kernel(const float* __restrict__ p2,
                              const float* __restrict__ poly,
                              _Float16* __restrict__ inp) {
    int bq = blockIdx.x;          // 0..NQ-1
    int c  = threadIdx.x;         // 0..255
    int b  = bq / QPTS;

    float cx = poly[bq * 2 + 0];
    float cy = poly[bq * 2 + 1];
    float ix = (cx * 2.f * WW - 1.f) * 0.5f;
    float iy = (cy * 2.f * HH - 1.f) * 0.5f;
    ix = fminf(fmaxf(ix, 0.f), (float)(WW - 1));
    iy = fminf(fmaxf(iy, 0.f), (float)(HH - 1));
    float x0f = floorf(ix), y0f = floorf(iy);
    float wx = ix - x0f,    wy = iy - y0f;
    int x0 = (int)x0f, y0 = (int)y0f;
    int x1 = (x0 + 1 < WW) ? x0 + 1 : WW - 1;
    int y1 = (y0 + 1 < HH) ? y0 + 1 : HH - 1;

    const float* fb = p2 + ((size_t)b * FDIM + c) * (size_t)(HH * WW);
    float f00 = fb[y0 * WW + x0];
    float f01 = fb[y0 * WW + x1];
    float f10 = fb[y1 * WW + x0];
    float f11 = fb[y1 * WW + x1];
    float top = f00 * (1.f - wx) + f01 * wx;
    float bot = f10 * (1.f - wx) + f11 * wx;

    _Float16* row = inp + (size_t)bq * 288;
    row[c] = (_Float16)(top * (1.f - wy) + bot * wy);
    if (c == 0) { row[256] = (_Float16)cx; row[257] = (_Float16)cy; }
    if (c >= 2 && c < 32) row[256 + c] = (_Float16)0.f;   // pad cols 258..287
}

// polygons = clip(polygons + clip(disp*SCALE, -2S, 2S), 0, 1)
__global__ void update_kernel(float* __restrict__ poly,
                              const float* __restrict__ disp, int n) {
    int i = blockIdx.x * blockDim.x + threadIdx.x;
    if (i >= n) return;
    float d = disp[i] * SCALEF;
    d = fminf(fmaxf(d, -2.f * SCALEF), 2.f * SCALEF);
    float p = poly[i] + d;
    poly[i] = fminf(fmaxf(p, 0.f), 1.f);
}

__global__ void copy_kernel(const float* __restrict__ src, float* __restrict__ dst, int n) {
    int i = blockIdx.x * blockDim.x + threadIdx.x;
    if (i < n) dst[i] = src[i];
}

// ---------------------------------------------------------------------------
// Host orchestration (graph-capture safe).
// ---------------------------------------------------------------------------
extern "C" void kernel_launch(void* const* d_in, const int* in_sizes, int n_in,
                              void* d_out, int out_size, void* d_ws, size_t ws_size,
                              hipStream_t stream) {
    (void)in_sizes; (void)n_in; (void)out_size; (void)ws_size;

    const float* p2  = (const float*)d_in[0];
    const float* p4  = (const float*)d_in[1];
    const float* iw1 = (const float*)d_in[2];  const float* ib1 = (const float*)d_in[3];
    const float* iw2 = (const float*)d_in[4];  const float* ib2 = (const float*)d_in[5];
    const float* iw3 = (const float*)d_in[6];  const float* ib3 = (const float*)d_in[7];
    const float* rw1 = (const float*)d_in[8];  const float* rb1 = (const float*)d_in[9];
    const float* rw2 = (const float*)d_in[10]; const float* rb2 = (const float*)d_in[11];
    const float* rw3 = (const float*)d_in[12]; const float* rb3 = (const float*)d_in[13];
    const float* vw1 = (const float*)d_in[14]; const float* vb1 = (const float*)d_in[15];
    const float* vw2 = (const float*)d_in[16]; const float* vb2 = (const float*)d_in[17];

    // ---- workspace: f32 region, then 16B-aligned f16 region (~28.5 MB) ----
    float* poly = (float*)d_ws;                    // 8000
    float* dsp  = poly + 8000;                     // 8000
    _Float16* hb = (_Float16*)(dsp + 8000);        // 64000 B offset (16B aligned)
    _Float16* fW1  = hb;                 hb += (size_t)16384 * 512;  // K2=16384,N2=512
    _Float16* fW2  = hb;                 hb += (size_t)512 * 1024;
    _Float16* fW3  = hb;                 hb += (size_t)1024 * 2048;  // N 2000->2048
    _Float16* fRW1 = hb;                 hb += (size_t)288 * 256;    // K 258->288
    _Float16* fRW2 = hb;                 hb += (size_t)256 * 128;
    _Float16* fRW3 = hb;                 hb += (size_t)128 * 64;     // N 2->64
    _Float16* fVW1 = hb;                 hb += (size_t)128 * 128;    // K 100->128
    _Float16* fVW2 = hb;                 hb += (size_t)128 * 64;     // N 1->64
    _Float16* fFlat = hb;                hb += (size_t)16 * 16384;   // M 4->16
    _Float16* fH1  = hb;                 hb += (size_t)16 * 512;
    _Float16* fH2  = hb;                 hb += (size_t)16 * 1024;
    _Float16* fInp = hb;                 hb += (size_t)NQ * 288;
    _Float16* fR1  = hb;                 hb += (size_t)NQ * 256;
    _Float16* fR2  = hb;                 hb += (size_t)NQ * 128;
    _Float16* fPoly = hb;                hb += (size_t)80 * 128;     // K 100->128
    _Float16* fV1  = hb;                 hb += (size_t)80 * 128;

    float* out_poly = (float*)d_out;               // [8000]
    float* out_val  = out_poly + 8000;             // [80]
    float* out_init = out_val + 80;                // [8000]

    dim3 blk(128);
    #define CVT(src, dst, M_, K_, M2_, K2_) \
        cvt_pad_kernel<<<((M2_) * (K2_) + 255) / 256, 256, 0, stream>>>(src, dst, M_, K_, M2_, K2_)
    #define GGRID(M2_, N2_) dim3((N2_) / 64, (M2_) / 16)

    // 0) weight prep: f32 -> padded f16 (~2 us of HBM traffic)
    CVT(iw1, fW1, 16384, 512, 16384, 512);
    CVT(iw2, fW2,   512, 1024,  512, 1024);
    CVT(iw3, fW3,  1024, 2000, 1024, 2048);
    CVT(rw1, fRW1,  258,  256,  288,  256);
    CVT(rw2, fRW2,  256,  128,  256,  128);
    CVT(rw3, fRW3,  128,    2,  128,   64);
    CVT(vw1, fVW1,  100,  128,  128,  128);
    CVT(vw2, fVW2,  128,    1,  128,   64);

    // 1) pool -> f16 flat rows 0..3; zero pad rows 4..15
    pool_kernel<<<(NB * FDIM * 64 + 255) / 256, 256, 0, stream>>>(p4, fFlat);
    fill_zero_h<<<(12 * 16384 + 255) / 256, 256, 0, stream>>>(fFlat + (size_t)4 * 16384, 12 * 16384);

    // 2) Init MLP
    wmma_gemm_kernel<<<GGRID(16,  512), blk, 0, stream>>>(fFlat, fW1, ib1, nullptr, fH1, NB,  512, 16384,  512, ACT_RELU);
    wmma_gemm_kernel<<<GGRID(16, 1024), blk, 0, stream>>>(fH1,   fW2, ib2, nullptr, fH2, NB, 1024,   512, 1024, ACT_RELU);
    wmma_gemm_kernel<<<GGRID(16, 2048), blk, 0, stream>>>(fH2,   fW3, ib3, out_init, nullptr, NB, 2000, 1024, 2048, ACT_SIGMOID);
    copy_kernel<<<(8000 + 255) / 256, 256, 0, stream>>>(out_init, poly, 8000);

    // 3) Three refinement steps
    for (int s = 0; s < 3; ++s) {
        sample_kernel<<<NQ, FDIM, 0, stream>>>(p2, poly, fInp);
        wmma_gemm_kernel<<<GGRID(NQ, 256), blk, 0, stream>>>(fInp, fRW1, rb1, nullptr, fR1, NQ, 256, 288, 256, ACT_RELU);
        wmma_gemm_kernel<<<GGRID(NQ, 128), blk, 0, stream>>>(fR1,  fRW2, rb2, nullptr, fR2, NQ, 128, 256, 128, ACT_RELU);
        wmma_gemm_kernel<<<GGRID(NQ,  64), blk, 0, stream>>>(fR2,  fRW3, rb3, dsp, nullptr, NQ, 2, 128, 64, ACT_TANH);
        update_kernel<<<(8000 + 255) / 256, 256, 0, stream>>>(poly, dsp, 8000);
    }

    // 4) Validity head: poly [80,100] -> padded f16 [80,128]
    CVT(poly, fPoly, 80, 100, 80, 128);
    wmma_gemm_kernel<<<GGRID(80, 128), blk, 0, stream>>>(fPoly, fVW1, vb1, nullptr, fV1, 80, 128, 128, 128, ACT_RELU);
    wmma_gemm_kernel<<<GGRID(80,  64), blk, 0, stream>>>(fV1,   fVW2, vb2, out_val, nullptr, 80, 1, 128, 64, ACT_SIGMOID);

    // 5) Final polygons
    copy_kernel<<<(8000 + 255) / 256, 256, 0, stream>>>(poly, out_poly, 8000);
}
